// TransformerBlock_24575802868263
// MI455X (gfx1250) — compile-verified, matrix operand order checked by reading, not run on hardware
//
#include <hip/hip_runtime.h>
#include <cstdint>
#include <cstddef>

// Problem constants (match reference).
#define BDIM  16
#define SDIM  1024
#define CDIM  512          // DIN == DOUT == H*DH
#define HDIM  8
#define DHD   64
#define MROWS (BDIM * SDIM)

typedef __attribute__((ext_vector_type(16))) __bf16 v16bf;
typedef __attribute__((ext_vector_type(8)))  float  v8f;

union FragBF {
    v16bf v;
    uint4 q[2];
    unsigned short s[16];
};

__device__ inline v8f vzero8() {
    v8f v = {0.f, 0.f, 0.f, 0.f, 0.f, 0.f, 0.f, 0.f};
    return v;
}

// f32 -> bf16 round-to-nearest-even (branchless; finite data in this workload)
__device__ inline unsigned short f2bf(float f) {
    union { float f; unsigned u; } c; c.f = f;
    unsigned r = c.u + 0x7FFFu + ((c.u >> 16) & 1u);
    return (unsigned short)(r >> 16);
}

// ---------------------------------------------------------------------------
// Kernel 0: f32 -> bf16 bulk convert, 4 elements/thread (b128 in, b64 out)
// ---------------------------------------------------------------------------
__global__ void k_cvt_bf16(const float* __restrict__ in,
                           unsigned short* __restrict__ out, int n4) {
    int i = blockIdx.x * blockDim.x + threadIdx.x;
    if (i < n4) {
        float4 f = ((const float4*)in)[i];
        union { unsigned short s[4]; uint2 v; } o;
        o.s[0] = f2bf(f.x); o.s[1] = f2bf(f.y);
        o.s[2] = f2bf(f.z); o.s[3] = f2bf(f.w);
        *(uint2*)(out + (size_t)i * 4) = o.v;
    }
}

// ---------------------------------------------------------------------------
// Kernel 1: C[M,N] = A[M,K](bf16) * W[K,N](f32 -> bf16)
// Block: 256 threads (8 waves). Block tile 128x64, K-step 64.
// Wave w computes rows [16w,16w+16) x all 64 cols (4 WMMA accumulators);
// the 64x64 B tile is amortized across all 8 waves.
// ---------------------------------------------------------------------------
template <bool STORE_BF16>
__global__ __launch_bounds__(256)
void k_gemm_bf16(const unsigned short* __restrict__ A,
                 const float* __restrict__ W,
                 void* __restrict__ Cout, int Mdim, int Ndim, int Kdim) {
    __shared__ unsigned short atile[128 * 64];       // [m][k], stride 64
    __shared__ unsigned short btile[64 * 64];        // transposed: [n][k]

    const int tid  = threadIdx.x;
    const int wave = tid >> 5;
    const int lane = tid & 31;
    const int hh   = lane >> 4;          // 0: lanes 0-15, 1: lanes 16-31
    const int l15  = lane & 15;
    const int koff = hh ? 8 : 0;         // ISA 16-bit A/B fragment K sub-offset
    const int m0   = blockIdx.y * 128;
    const int n0   = blockIdx.x * 64;

    v8f acc[4];
#pragma unroll
    for (int j = 0; j < 4; ++j) acc[j] = vzero8();

    for (int k0 = 0; k0 < Kdim; k0 += 64) {
        __syncthreads();
        // --- stage A tile: 128x64 bf16, coalesced 16B loads ---
#pragma unroll
        for (int i = 0; i < 4; ++i) {
            int li = tid + 256 * i;                  // 0..1023
            int r  = li >> 3;                        // 0..127
            int c  = (li & 7) * 8;                   // 0..56
            *(uint4*)(atile + r * 64 + c) =
                *(const uint4*)(A + (size_t)(m0 + r) * Kdim + k0 + c);
        }
        // --- stage W tile: 64x64 f32 load, bf16 convert, transpose [n][k] ---
#pragma unroll
        for (int i = 0; i < 4; ++i) {
            int li = tid + 256 * i;                  // 0..1023
            int kr = li >> 4;                        // 0..63
            int c  = (li & 15) * 4;                  // 0..60
            float4 w4 = *(const float4*)(W + (size_t)(k0 + kr) * Ndim + n0 + c);
            btile[(c + 0) * 64 + kr] = f2bf(w4.x);
            btile[(c + 1) * 64 + kr] = f2bf(w4.y);
            btile[(c + 2) * 64 + kr] = f2bf(w4.z);
            btile[(c + 3) * 64 + kr] = f2bf(w4.w);
        }
        __syncthreads();

        // Speculative prefetch of next A tile (global_prefetch_b8)
        if (k0 + 64 < Kdim) {
            __builtin_prefetch(A + (size_t)(m0 + (tid >> 1)) * Kdim + k0 + 64, 0, 1);
        }

        // Two K=32 windows per staged K=64
#pragma unroll
        for (int t = 0; t < 2; ++t) {
            FragBF a;
            {
                const unsigned short* ar =
                    atile + ((wave << 4) + l15) * 64 + t * 32;
                a.q[0] = *(const uint4*)(ar + koff);
                a.q[1] = *(const uint4*)(ar + 16 + koff);
            }
#pragma unroll
            for (int j = 0; j < 4; ++j) {
                FragBF b;
                const unsigned short* br =
                    btile + ((j << 4) + l15) * 64 + t * 32;
                b.q[0] = *(const uint4*)(br + koff);
                b.q[1] = *(const uint4*)(br + 16 + koff);
                acc[j] = __builtin_amdgcn_wmma_f32_16x16x32_bf16(
                    false, a.v, false, b.v, (short)0, acc[j], false, false);
            }
        }
    }

    // Epilogue: C layout -> lane l element r holds (m = r + 8*hh, n = l15)
    const int mbase = m0 + (wave << 4) + (hh ? 8 : 0);
#pragma unroll
    for (int j = 0; j < 4; ++j) {
        const int n = n0 + (j << 4) + l15;
#pragma unroll
        for (int r = 0; r < 8; ++r) {
            const size_t idx = (size_t)(mbase + r) * Ndim + n;
            if (STORE_BF16) ((unsigned short*)Cout)[idx] = f2bf(acc[j][r]);
            else            ((float*)Cout)[idx] = acc[j][r];
        }
    }
}

// ---------------------------------------------------------------------------
// Kernel 2: fused flash attention. Block = 128 threads = 4 waves; block owns
// (b, h, 64 q-rows); each wave owns 16 q-rows resident in A-fragments.
// KV tiled by 32 rows; K staged natural [kv][dh], V staged TRANSPOSED
// [dh][kv] so P*V B-fragments are two ds_load_b128 each.
// ---------------------------------------------------------------------------
__global__ __launch_bounds__(128)
void k_attn(const unsigned short* __restrict__ Q,
            const unsigned short* __restrict__ K,
            const unsigned short* __restrict__ V,
            unsigned short* __restrict__ Z) {
    __shared__ unsigned short ktile[32 * 64];     // [kv][dh]
    __shared__ unsigned short vtile[64 * 32];     // transposed: [dh][kv]
    __shared__ unsigned short ptile[4][16 * 32];  // per-wave P tile [m][kv]

    const int tid  = threadIdx.x;
    const int wave = tid >> 5;
    const int lane = tid & 31;
    const int hh   = lane >> 4;
    const int l15  = lane & 15;
    const int koff = hh ? 8 : 0;
    const int h    = blockIdx.y;
    const int b    = blockIdx.z;
    const int s0   = blockIdx.x * 64 + wave * 16;      // this wave's q rows
    const int RS   = HDIM * DHD;                       // 512 row stride

    // Resident Q fragments: 2 K-windows over dh (0..31, 32..63)
    FragBF aq[2];
    {
        const unsigned short* qp =
            Q + (size_t)(b * SDIM + s0 + l15) * RS + h * DHD;
#pragma unroll
        for (int t = 0; t < 2; ++t) {
            aq[t].q[0] = *(const uint4*)(qp + t * 32 + koff);
            aq[t].q[1] = *(const uint4*)(qp + t * 32 + 16 + koff);
        }
    }

    v8f oacc[4];
#pragma unroll
    for (int j = 0; j < 4; ++j) oacc[j] = vzero8();
    float mrow[8], lrow[8];
#pragma unroll
    for (int r = 0; r < 8; ++r) { mrow[r] = -1e30f; lrow[r] = 0.f; }

    const float sc = 0.125f;  // 1/sqrt(DH)

    for (int kv0 = 0; kv0 < SDIM; kv0 += 32) {
        __syncthreads();
        // stage K (natural) and V (transposed) tiles
#pragma unroll
        for (int i = 0; i < 2; ++i) {
            int li = tid + 128 * i;                 // 0..255
            int r  = li >> 3;                       // 0..31  (kv row)
            int c  = (li & 7) * 8;                  // 0..56  (dh col)
            const size_t g = (size_t)(b * SDIM + kv0 + r) * RS + h * DHD + c;
            *(uint4*)(ktile + r * 64 + c) = *(const uint4*)(K + g);
            uint4 vv = *(const uint4*)(V + g);
            const unsigned short* vs = (const unsigned short*)&vv;
#pragma unroll
            for (int e = 0; e < 8; ++e) vtile[(c + e) * 32 + r] = vs[e];
        }
        __syncthreads();

        // scores: S = Q * K^T  (two 16-col score tiles, two dh K-windows)
        v8f sA = vzero8(), sB = vzero8();
#pragma unroll
        for (int t = 0; t < 2; ++t) {
            FragBF bk0, bk1;
            const unsigned short* kr0 = ktile + l15 * 64 + t * 32;
            const unsigned short* kr1 = ktile + (16 + l15) * 64 + t * 32;
            bk0.q[0] = *(const uint4*)(kr0 + koff);
            bk0.q[1] = *(const uint4*)(kr0 + 16 + koff);
            bk1.q[0] = *(const uint4*)(kr1 + koff);
            bk1.q[1] = *(const uint4*)(kr1 + 16 + koff);
            sA = __builtin_amdgcn_wmma_f32_16x16x32_bf16(
                false, aq[t].v, false, bk0.v, (short)0, sA, false, false);
            sB = __builtin_amdgcn_wmma_f32_16x16x32_bf16(
                false, aq[t].v, false, bk1.v, (short)0, sB, false, false);
        }

        // online softmax over 32 new columns (row lives in element r of 16 lanes)
        unsigned short* pw = ptile[wave];
#pragma unroll
        for (int r = 0; r < 8; ++r) {
            float x0 = sA[r] * sc, x1 = sB[r] * sc;
            float mx = fmaxf(x0, x1);
#pragma unroll
            for (int off = 1; off < 16; off <<= 1)
                mx = fmaxf(mx, __shfl_xor(mx, off, 16));
            float mnew  = fmaxf(mrow[r], mx);
            float scale = __expf(mrow[r] - mnew);
            float p0 = __expf(x0 - mnew);
            float p1 = __expf(x1 - mnew);
            float ps = p0 + p1;
#pragma unroll
            for (int off = 1; off < 16; off <<= 1)
                ps += __shfl_xor(ps, off, 16);
            lrow[r] = lrow[r] * scale + ps;
            mrow[r] = mnew;
#pragma unroll
            for (int j = 0; j < 4; ++j) oacc[j][r] *= scale;
            const int m = r + (hh ? 8 : 0);
            pw[m * 32 + l15]      = f2bf(p0);
            pw[m * 32 + 16 + l15] = f2bf(p1);
        }

        // O += P * V (P re-shaped C-layout -> A-fragment via per-wave LDS;
        //             V fragments are contiguous b128 reads of vtile rows)
        FragBF ap;
        {
            const unsigned short* pr = pw + l15 * 32;
            ap.q[0] = *(const uint4*)(pr + koff);
            ap.q[1] = *(const uint4*)(pr + 16 + koff);
        }
#pragma unroll
        for (int j = 0; j < 4; ++j) {
            FragBF bv;
            const unsigned short* vr = vtile + ((j << 4) + l15) * 32;
            bv.q[0] = *(const uint4*)(vr + koff);
            bv.q[1] = *(const uint4*)(vr + 16 + koff);
            oacc[j] = __builtin_amdgcn_wmma_f32_16x16x32_bf16(
                false, ap.v, false, bv.v, (short)0, oacc[j], false, false);
        }
    }

    // normalize by row sum and store bf16 Z[b, s, h, dh]
    const int mbase = s0 + (hh ? 8 : 0);
#pragma unroll
    for (int r = 0; r < 8; ++r) {
        const float rinv = 1.0f / lrow[r];
#pragma unroll
        for (int j = 0; j < 4; ++j) {
            const int dh = (j << 4) + l15;
            Z[(size_t)(b * SDIM + mbase + r) * RS + h * DHD + dh] =
                f2bf(oacc[j][r] * rinv);
        }
    }
}

// ---------------------------------------------------------------------------
// Kernel 3: per-row bias + LayerNorm (+residual) + ReLU. One 256-thread block
// per row of 512; 2 elements/thread; LDS tree reductions.
// ---------------------------------------------------------------------------
template <bool HAS_RES, bool OUT_BF16>
__global__ __launch_bounds__(256)
void k_ln(const float* __restrict__ Y, const float* __restrict__ bias,
          const float* __restrict__ res, const float* __restrict__ g,
          const float* __restrict__ be, void* __restrict__ Out) {
    __shared__ float red[256];
    const int row = blockIdx.x;
    const int tid = threadIdx.x;
    const float* y = Y + (size_t)row * CDIM;

    float v0 = y[tid] + bias[tid];
    float v1 = y[tid + 256] + bias[tid + 256];

    red[tid] = v0 + v1;
#pragma unroll
    for (int st = 128; st > 0; st >>= 1) {
        __syncthreads();
        if (tid < st) red[tid] += red[tid + st];
    }
    __syncthreads();
    const float mu = red[0] * (1.0f / CDIM);
    __syncthreads();

    const float d0 = v0 - mu, d1 = v1 - mu;
    red[tid] = d0 * d0 + d1 * d1;
#pragma unroll
    for (int st = 128; st > 0; st >>= 1) {
        __syncthreads();
        if (tid < st) red[tid] += red[tid + st];
    }
    __syncthreads();
    const float rstd = rsqrtf(red[0] * (1.0f / CDIM) + 1e-5f);

    float o0 = d0 * rstd * g[tid] + be[tid];
    float o1 = d1 * rstd * g[tid + 256] + be[tid + 256];
    if (HAS_RES) {
        o0 += res[(size_t)row * CDIM + tid];
        o1 += res[(size_t)row * CDIM + tid + 256];
    }
    o0 = fmaxf(o0, 0.f);
    o1 = fmaxf(o1, 0.f);
    if (OUT_BF16) {
        unsigned short* o = (unsigned short*)Out + (size_t)row * CDIM;
        o[tid] = f2bf(o0);
        o[tid + 256] = f2bf(o1);
    } else {
        float* o = (float*)Out + (size_t)row * CDIM;
        o[tid] = o0;
        o[tid + 256] = o1;
    }
}

// ---------------------------------------------------------------------------
extern "C" void kernel_launch(void* const* d_in, const int* in_sizes, int n_in,
                              void* d_out, int out_size, void* d_ws, size_t ws_size,
                              hipStream_t stream) {
    const float* x     = (const float*)d_in[0];
    const float* wq    = (const float*)d_in[1];
    const float* wk    = (const float*)d_in[2];
    const float* wv    = (const float*)d_in[3];
    const float* w_att = (const float*)d_in[4];
    const float* b_att = (const float*)d_in[5];
    const float* g1    = (const float*)d_in[6];
    const float* be1   = (const float*)d_in[7];
    const float* w_ff  = (const float*)d_in[8];
    const float* b_ff  = (const float*)d_in[9];
    const float* g2    = (const float*)d_in[10];
    const float* be2   = (const float*)d_in[11];

    const size_t nElem = (size_t)MROWS * CDIM;  // 8.4M elements
    unsigned short* xb = (unsigned short*)d_ws; // also reused as z2 (bf16)
    unsigned short* qb = xb + nElem;
    unsigned short* kb = qb + nElem;
    unsigned short* vb = kb + nElem;
    unsigned short* zb = vb + nElem;
    float* yf = (float*)((char*)d_ws + nElem * 2 * 5);

    // 1) x -> bf16 (vectorized x4)
    const int n4 = (int)(nElem / 4);
    k_cvt_bf16<<<dim3((n4 + 255) / 256), dim3(256), 0, stream>>>(x, xb, n4);

    // 2) QKV projections (bf16 WMMA GEMMs, bf16 outputs)
    const dim3 gg(CDIM / 64, MROWS / 128);
    k_gemm_bf16<true><<<gg, 256, 0, stream>>>(xb, wq, qb, MROWS, CDIM, CDIM);
    k_gemm_bf16<true><<<gg, 256, 0, stream>>>(xb, wk, kb, MROWS, CDIM, CDIM);
    k_gemm_bf16<true><<<gg, 256, 0, stream>>>(xb, wv, vb, MROWS, CDIM, CDIM);

    // 3) fused flash attention -> zb (bf16)
    k_attn<<<dim3(SDIM / 64, HDIM, BDIM), 128, 0, stream>>>(qb, kb, vb, zb);

    // 4) output projection -> yf (f32)
    k_gemm_bf16<false><<<gg, 256, 0, stream>>>(zb, w_att, yf, MROWS, CDIM, CDIM);

    // 5) z2 = relu(x + LN(yf + b_att)) -> bf16 (reuse xb region)
    k_ln<true, true><<<dim3(MROWS), dim3(256), 0, stream>>>(
        yf, b_att, x, g1, be1, xb);

    // 6) FF GEMM -> yf (f32)
    k_gemm_bf16<false><<<gg, 256, 0, stream>>>(xb, w_ff, yf, MROWS, CDIM, CDIM);

    // 7) out = relu(LN(yf + b_ff)) -> f32 d_out
    k_ln<false, false><<<dim3(MROWS), dim3(256), 0, stream>>>(
        yf, b_ff, nullptr, g2, be2, d_out);
}
